// TransformerModel_58059367908141
// MI455X (gfx1250) — compile-verified, matrix operand order checked by reading, not run on hardware
//
#include <hip/hip_runtime.h>
#include <math.h>
#include <stdint.h>

typedef __bf16 bf16;
typedef __bf16 v16bf __attribute__((ext_vector_type(16)));
typedef float  v8f   __attribute__((ext_vector_type(8)));
typedef int    i4v   __attribute__((vector_size(16)));
typedef __attribute__((address_space(1))) i4v* gptr_b128;
typedef __attribute__((address_space(3))) i4v* lptr_b128;

static constexpr int S_  = 2048;
static constexpr int D_  = 1024;
static constexpr int H_  = 16;
static constexpr int HD_ = 64;
static constexpr int F_  = 4096;
static constexpr int L_  = 4;
static constexpr int V_  = 32000;
static constexpr float EPS_ = 1e-6f;

// ---------------- fragment helpers (wave32 WMMA layouts, ISA 7.12.2) ----------------
union FragB16 { v16bf v; unsigned int u[8]; };
union FragF32 { v8f v; float f[8]; };

__device__ __forceinline__ int kpat(int p, int lk8) {
  return (p < 4) ? (lk8 + 2 * p) : (16 + lk8 + 2 * (p - 4));
}

// ---------------- async global->LDS copy (GLOBAL_LOAD_ASYNC_TO_LDS_B128) ----------------
__device__ __forceinline__ void async_b128(const void* g, void* l) {
  unsigned long long ga = (unsigned long long)(uintptr_t)g;
  unsigned la = (unsigned)(uintptr_t)l;  // flat shared addr low 32 bits == LDS offset (ISA 10.2)
#if __has_builtin(__builtin_amdgcn_global_load_async_to_lds_b128)
  __builtin_amdgcn_global_load_async_to_lds_b128((gptr_b128)ga, (lptr_b128)la, 0, 0);
#else
  asm volatile("global_load_async_to_lds_b128 %0, %1, off"
               :: "v"((lptr_b128)la), "v"(ga)
               : "memory");
#endif
}

__device__ __forceinline__ void wait_async0() {
#if __has_builtin(__builtin_amdgcn_s_wait_asynccnt)
  __builtin_amdgcn_s_wait_asynccnt(0);
#else
  asm volatile("s_wait_asynccnt 0x0" ::: "memory");
#endif
}

// ---------------- utility kernels ----------------
__global__ __launch_bounds__(256) void cvt_f32_bf16_kernel(
    const float* __restrict__ src, bf16* __restrict__ dst, long n) {
  long i = (long)blockIdx.x * blockDim.x + threadIdx.x;
  long stride = (long)gridDim.x * blockDim.x;
  for (; i < n; i += stride) dst[i] = (bf16)src[i];
}

// src: K x N fp32 -> dst: N x K bf16 (one-time weight pre-transpose)
__global__ __launch_bounds__(256) void cvt_transpose_kernel(
    const float* __restrict__ src, bf16* __restrict__ dst, int K, int N) {
  __shared__ float t[32][33];
  int kb = blockIdx.y * 32, nb = blockIdx.x * 32;
  int tx = threadIdx.x & 31, ty = threadIdx.x >> 5;
  for (int i = ty; i < 32; i += 8) t[i][tx] = src[(long)(kb + i) * N + nb + tx];
  __syncthreads();
  for (int i = ty; i < 32; i += 8) dst[(long)(nb + i) * K + kb + tx] = (bf16)t[tx][i];
}

__global__ __launch_bounds__(256) void embed_kernel(
    const int* __restrict__ ids, const float* __restrict__ tok,
    const float* __restrict__ pos, float* __restrict__ h) {
  int s = blockIdx.x;
  int id = ids[s];
  for (int d = threadIdx.x; d < D_; d += 256)
    h[(long)s * D_ + d] = tok[(long)id * D_ + d] + pos[(long)s * D_ + d];
}

__global__ __launch_bounds__(256) void layernorm_kernel(
    const float* __restrict__ h, const float* __restrict__ g,
    const float* __restrict__ b, bf16* __restrict__ x) {
  __shared__ float rsum[256], rsq[256];
  const int s = blockIdx.x, t = threadIdx.x;
  const float* row = h + (long)s * D_;
  float v0 = row[t], v1 = row[t + 256], v2 = row[t + 512], v3 = row[t + 768];
  rsum[t] = v0 + v1 + v2 + v3;
  rsq[t]  = v0 * v0 + v1 * v1 + v2 * v2 + v3 * v3;
  __syncthreads();
  for (int off = 128; off > 0; off >>= 1) {
    if (t < off) { rsum[t] += rsum[t + off]; rsq[t] += rsq[t + off]; }
    __syncthreads();
  }
  float mean = rsum[0] * (1.0f / D_);
  float var  = rsq[0] * (1.0f / D_) - mean * mean;
  float rstd = rsqrtf(var + EPS_);
  bf16* xo = x + (long)s * D_;
  xo[t]       = (bf16)((v0 - mean) * rstd * g[t]       + b[t]);
  xo[t + 256] = (bf16)((v1 - mean) * rstd * g[t + 256] + b[t + 256]);
  xo[t + 512] = (bf16)((v2 - mean) * rstd * g[t + 512] + b[t + 512]);
  xo[t + 768] = (bf16)((v3 - mean) * rstd * g[t + 768] + b[t + 768]);
}

// ---------------- WMMA GEMM: C = A(MxK) @ Bt(NxK)^T + bias, fused epilogues ----------------
// Async double-buffered LDS pipeline: tile k+1 streams memory->LDS (ASYNCcnt)
// while WMMAs consume tile k.
static constexpr int EPI_QKV = 0;    // scatter bf16 into [head][S][64]
static constexpr int EPI_RESID = 1;  // f32 out = resid + acc + bias
static constexpr int EPI_GELU = 2;   // bf16 out = gelu(acc + bias)
static constexpr int EPI_F32 = 3;    // f32 out = acc (+ bias if non-null)

static constexpr int BM = 128, BN = 128, BK = 32;
static constexpr int LDT = BK + 8;       // 40 halves = 80B row stride (16B multiple)
static constexpr int TSZ = BM * LDT;     // one LDS tile

template <int EPI>
__global__ __launch_bounds__(256) void gemm_bf16_kernel(
    const bf16* __restrict__ A, const bf16* __restrict__ Bt,
    const float* __restrict__ bias, const float* __restrict__ resid,
    float* __restrict__ outf, bf16* __restrict__ outb, int M, int N, int K) {
  __shared__ bf16 sA[2 * TSZ];
  __shared__ bf16 sB[2 * TSZ];  // Bt tile: [n][k]
  const int tid = threadIdx.x;
  const int lane = tid & 31, w = tid >> 5;
  const int wy = w >> 2, wx = w & 3;  // 2x4 wave grid -> wave tile 64(M) x 32(N)
  const int m0 = blockIdx.y * BM, n0 = blockIdx.x * BN;
  const int lm = lane & 15, lk8 = (lane >> 4) * 8;

  FragF32 acc[4][2];
#pragma unroll
  for (int i = 0; i < 4; i++)
#pragma unroll
    for (int j = 0; j < 2; j++)
#pragma unroll
      for (int e = 0; e < 8; e++) acc[i][j].f[e] = 0.0f;

  // async-stage one 128x32 tile pair (512 b128 ops each, 2 per thread)
  auto stage = [&](int buf, int k0) {
#pragma unroll
    for (int j = 0; j < 2; j++) {
      int i = tid + j * 256;
      int r = i >> 2, c8 = (i & 3) * 8;
      async_b128(A + (long)(m0 + r) * K + k0 + c8, sA + buf * TSZ + r * LDT + c8);
      async_b128(Bt + (long)(n0 + r) * K + k0 + c8, sB + buf * TSZ + r * LDT + c8);
    }
  };

  stage(0, 0);
  wait_async0();
  __syncthreads();

  const int niter = K / BK;
  for (int it = 0; it < niter; it++) {
    const int cur = it & 1, nxt = cur ^ 1;
    if (it + 1 < niter) stage(nxt, (it + 1) * BK);  // overlap DMA with WMMAs

    FragB16 af[4], bfr[2];
#pragma unroll
    for (int mt = 0; mt < 4; mt++) {
      const bf16* base = sA + cur * TSZ + (wy * 64 + mt * 16 + lm) * LDT;
#pragma unroll
      for (int p = 0; p < 8; p++)
        af[mt].u[p] = *(const unsigned int*)(base + kpat(p, lk8));
    }
#pragma unroll
    for (int nt = 0; nt < 2; nt++) {
      const bf16* base = sB + cur * TSZ + (wx * 32 + nt * 16 + lm) * LDT;
#pragma unroll
      for (int p = 0; p < 8; p++)
        bfr[nt].u[p] = *(const unsigned int*)(base + kpat(p, lk8));
    }
#pragma unroll
    for (int mt = 0; mt < 4; mt++)
#pragma unroll
      for (int nt = 0; nt < 2; nt++)
        acc[mt][nt].v = __builtin_amdgcn_wmma_f32_16x16x32_bf16(
            false, af[mt].v, false, bfr[nt].v, (short)0, acc[mt][nt].v, false, false);

    wait_async0();   // next tile landed (own ASYNCcnt)
    __syncthreads(); // all waves done reading cur + done waiting
  }

  // epilogue (C/D layout: lane half picks M base, r = vgpr index, n = lane&15)
#pragma unroll
  for (int mt = 0; mt < 4; mt++)
#pragma unroll
    for (int nt = 0; nt < 2; nt++) {
      int gmB = m0 + wy * 64 + mt * 16 + lk8;
      int gn = n0 + wx * 32 + nt * 16 + lm;
      float bv = bias ? bias[gn] : 0.0f;
#pragma unroll
      for (int r = 0; r < 8; r++) {
        int gm = gmB + r;
        float val = acc[mt][nt].f[r] + bv;
        if (EPI == EPI_QKV) {
          outb[((long)(gn >> 6) * S_ + gm) * HD_ + (gn & 63)] = (bf16)val;
        } else if (EPI == EPI_RESID) {
          outf[(long)gm * N + gn] = resid[(long)gm * N + gn] + val;
        } else if (EPI == EPI_GELU) {
          float ge = 0.5f * val * (1.0f + erff(val * 0.70710678118654752f));
          outb[(long)gm * N + gn] = (bf16)ge;
        } else {
          outf[(long)gm * N + gn] = val;
        }
      }
    }
}

// ---------------- flash attention: per block = (head, 128 q rows) ----------------
static constexpr int BQ = 128, BKV = 128;
static constexpr int LDQ = HD_ + 8;   // 72 halves = 144B (16B multiple)
static constexpr int LDP = BKV + 8;   // 136
static constexpr int LDVT = BKV + 8;  // 136

__global__ __launch_bounds__(256) void attention_kernel(
    const bf16* __restrict__ Q, const bf16* __restrict__ Kb,
    const bf16* __restrict__ Vb, bf16* __restrict__ O) {
  __shared__ bf16 sQ[BQ * LDQ];
  __shared__ bf16 sK[BKV * LDQ];
  __shared__ bf16 sVt[HD_ * LDVT];   // V transposed: [hd][kv]
  __shared__ bf16 sP[BQ * LDP];      // probs, per-wave 16-row slabs

  const int h = blockIdx.y, qb = blockIdx.x * BQ;
  const int tid = threadIdx.x, lane = tid & 31, w = tid >> 5;
  const int lm = lane & 15, lk8 = (lane >> 4) * 8;
  const bf16* Qh = Q + (long)h * S_ * HD_;
  const bf16* Kh = Kb + (long)h * S_ * HD_;
  const bf16* Vh = Vb + (long)h * S_ * HD_;

  // async-stage Q block: 128 rows x 64 halves = 1024 b128 ops, 4 per thread
#pragma unroll
  for (int j = 0; j < 4; j++) {
    int i = tid + j * 256;
    int r = i >> 3, c8 = (i & 7) * 8;
    async_b128(Qh + (long)(qb + r) * HD_ + c8, sQ + r * LDQ + c8);
  }
  wait_async0();
  __syncthreads();

  FragB16 qf[2];  // wave's 16 q rows x 64 head-dim, two K=32 fragments
#pragma unroll
  for (int kk = 0; kk < 2; kk++)
#pragma unroll
    for (int p = 0; p < 8; p++)
      qf[kk].u[p] =
          *(const unsigned int*)(sQ + (w * 16 + lm) * LDQ + kk * 32 + kpat(p, lk8));

  FragF32 oacc[4];
#pragma unroll
  for (int t = 0; t < 4; t++)
#pragma unroll
    for (int e = 0; e < 8; e++) oacc[t].f[e] = 0.0f;
  float mrow[8], lrow[8];
#pragma unroll
  for (int r = 0; r < 8; r++) { mrow[r] = -3.0e38f; lrow[r] = 0.0f; }

  const int nblocks = blockIdx.x + 1;  // causal: only kv-blocks <= q-block
  for (int jb = 0; jb < nblocks; jb++) {
    const int kv0 = jb * BKV;
    // async K rows; Vt transpose goes through VGPR->ds while K streams
#pragma unroll
    for (int j = 0; j < 4; j++) {
      int i = tid + j * 256;
      int r = i >> 3, c8 = (i & 7) * 8;
      async_b128(Kh + (long)(kv0 + r) * HD_ + c8, sK + r * LDQ + c8);
    }
    for (int i = tid; i < BKV * HD_; i += 256) {
      int d = i & (HD_ - 1), r = i >> 6;
      sVt[d * LDVT + r] = Vh[(long)(kv0 + r) * HD_ + d];
    }
    wait_async0();
    __syncthreads();

    // scores: 16 q rows x 128 kv cols = 8 tiles, 2 WMMAs each over head dim
    FragF32 st[8];
#pragma unroll
    for (int nt = 0; nt < 8; nt++) {
#pragma unroll
      for (int e = 0; e < 8; e++) st[nt].f[e] = 0.0f;
#pragma unroll
      for (int kk = 0; kk < 2; kk++) {
        FragB16 kf;
#pragma unroll
        for (int p = 0; p < 8; p++)
          kf.u[p] = *(const unsigned int*)(sK + (nt * 16 + lm) * LDQ + kk * 32 +
                                           kpat(p, lk8));
        st[nt].v = __builtin_amdgcn_wmma_f32_16x16x32_bf16(
            false, qf[kk].v, false, kf.v, (short)0, st[nt].v, false, false);
      }
    }
    // scale + causal mask
    const int growB = qb + w * 16 + lk8;
#pragma unroll
    for (int nt = 0; nt < 8; nt++) {
      int gcol = kv0 + nt * 16 + lm;
#pragma unroll
      for (int r = 0; r < 8; r++) {
        float sv = st[nt].f[r] * 0.125f;  // 1/sqrt(64)
        if (gcol > growB + r) sv = -3.0e38f;
        st[nt].f[r] = sv;
      }
    }
    // online softmax (row reductions across the 16 lanes holding each row)
#pragma unroll
    for (int r = 0; r < 8; r++) {
      float mx = st[0].f[r];
#pragma unroll
      for (int nt = 1; nt < 8; nt++) mx = fmaxf(mx, st[nt].f[r]);
#pragma unroll
      for (int off = 1; off < 16; off <<= 1) mx = fmaxf(mx, __shfl_xor(mx, off, 32));
      float mnew = fmaxf(mrow[r], mx);
      float alpha = __expf(mrow[r] - mnew);
      mrow[r] = mnew;
#pragma unroll
      for (int t = 0; t < 4; t++) oacc[t].f[r] *= alpha;
      float rs = 0.0f;
#pragma unroll
      for (int nt = 0; nt < 8; nt++) {
        float p = __expf(st[nt].f[r] - mnew);
        st[nt].f[r] = p;
        rs += p;
      }
#pragma unroll
      for (int off = 1; off < 16; off <<= 1) rs += __shfl_xor(rs, off, 32);
      lrow[r] = lrow[r] * alpha + rs;
    }
    // stash P (bf16) in LDS, then PV via WMMA
#pragma unroll
    for (int nt = 0; nt < 8; nt++)
#pragma unroll
      for (int r = 0; r < 8; r++)
        sP[(w * 16 + lk8 + r) * LDP + nt * 16 + lm] = (bf16)st[nt].f[r];
    __syncthreads();
#pragma unroll
    for (int ks = 0; ks < 4; ks++) {
      FragB16 pf;
#pragma unroll
      for (int p = 0; p < 8; p++)
        pf.u[p] = *(const unsigned int*)(sP + (w * 16 + lm) * LDP + ks * 32 +
                                         kpat(p, lk8));
#pragma unroll
      for (int t = 0; t < 4; t++) {
        FragB16 vf;
#pragma unroll
        for (int p = 0; p < 8; p++)
          vf.u[p] = *(const unsigned int*)(sVt + (t * 16 + lm) * LDVT + ks * 32 +
                                           kpat(p, lk8));
        oacc[t].v = __builtin_amdgcn_wmma_f32_16x16x32_bf16(
            false, pf.v, false, vf.v, (short)0, oacc[t].v, false, false);
      }
    }
    __syncthreads();
  }

  // normalize + write attn output interleaved back to (S, D) bf16
#pragma unroll
  for (int t = 0; t < 4; t++)
#pragma unroll
    for (int r = 0; r < 8; r++) {
      int grow = qb + w * 16 + lk8 + r;
      int gcol = t * 16 + lm;
      O[(long)grow * D_ + h * HD_ + gcol] = (bf16)(oacc[t].f[r] / lrow[r]);
    }
}

// ---------------- host orchestration ----------------
extern "C" void kernel_launch(void* const* d_in, const int* in_sizes, int n_in,
                              void* d_out, int out_size, void* d_ws, size_t ws_size,
                              hipStream_t stream) {
  (void)in_sizes; (void)n_in; (void)out_size; (void)ws_size;
  const int*   ids  = (const int*)d_in[0];
  const float* tok  = (const float*)d_in[1];
  const float* pos  = (const float*)d_in[2];
  const float* Wq   = (const float*)d_in[3];
  const float* bq   = (const float*)d_in[4];
  const float* Wk   = (const float*)d_in[5];
  const float* bk   = (const float*)d_in[6];
  const float* Wv   = (const float*)d_in[7];
  const float* bv   = (const float*)d_in[8];
  const float* Wo   = (const float*)d_in[9];
  const float* bo   = (const float*)d_in[10];
  const float* ln1g = (const float*)d_in[11];
  const float* ln1b = (const float*)d_in[12];
  const float* ln2g = (const float*)d_in[13];
  const float* ln2b = (const float*)d_in[14];
  const float* W1   = (const float*)d_in[15];
  const float* b1   = (const float*)d_in[16];
  const float* W2   = (const float*)d_in[17];
  const float* b2   = (const float*)d_in[18];
  const float* fg   = (const float*)d_in[19];
  const float* fb   = (const float*)d_in[20];
  float* logits = (float*)d_out;

  char* ws = (char*)d_ws;
  size_t off = 0;
  auto carve = [&](size_t bytes) {
    void* p = ws + off;
    off = (off + bytes + 255) & ~(size_t)255;
    return p;
  };
  bf16* WqB = (bf16*)carve((size_t)L_ * D_ * D_ * 2);  // transposed: N x K
  bf16* WkB = (bf16*)carve((size_t)L_ * D_ * D_ * 2);
  bf16* WvB = (bf16*)carve((size_t)L_ * D_ * D_ * 2);
  bf16* WoB = (bf16*)carve((size_t)L_ * D_ * D_ * 2);
  bf16* W1B = (bf16*)carve((size_t)L_ * D_ * F_ * 2);  // F x D per layer
  bf16* W2B = (bf16*)carve((size_t)L_ * F_ * D_ * 2);  // D x F per layer
  bf16* tokB = (bf16*)carve((size_t)V_ * D_ * 2);      // already N x K
  float* hB = (float*)carve((size_t)S_ * D_ * 4);
  bf16* xB  = (bf16*)carve((size_t)S_ * D_ * 2);
  bf16* qB  = (bf16*)carve((size_t)S_ * D_ * 2);
  bf16* kB  = (bf16*)carve((size_t)S_ * D_ * 2);
  bf16* vB  = (bf16*)carve((size_t)S_ * D_ * 2);
  bf16* aoB = (bf16*)carve((size_t)S_ * D_ * 2);
  bf16* ffB = (bf16*)carve((size_t)S_ * F_ * 2);

  auto cvtT = [&](const float* s, bf16* d, int K, int N) {  // KxN f32 -> NxK bf16
    cvt_transpose_kernel<<<dim3(N / 32, K / 32), dim3(256), 0, stream>>>(s, d, K, N);
  };
  for (int l = 0; l < L_; l++) {
    cvtT(Wq + (long)l * D_ * D_, WqB + (long)l * D_ * D_, D_, D_);
    cvtT(Wk + (long)l * D_ * D_, WkB + (long)l * D_ * D_, D_, D_);
    cvtT(Wv + (long)l * D_ * D_, WvB + (long)l * D_ * D_, D_, D_);
    cvtT(Wo + (long)l * D_ * D_, WoB + (long)l * D_ * D_, D_, D_);
    cvtT(W1 + (long)l * D_ * F_, W1B + (long)l * D_ * F_, D_, F_);
    cvtT(W2 + (long)l * F_ * D_, W2B + (long)l * F_ * D_, F_, D_);
  }
  {
    long n = (long)V_ * D_;
    cvt_f32_bf16_kernel<<<dim3(16384), dim3(256), 0, stream>>>(tok, tokB, n);
  }

  embed_kernel<<<dim3(S_), dim3(256), 0, stream>>>(ids, tok, pos, hB);

  const dim3 blk(256);
  const dim3 gDD(D_ / BN, S_ / BM);
  const dim3 gDF(F_ / BN, S_ / BM);
  for (int l = 0; l < L_; l++) {
    layernorm_kernel<<<dim3(S_), blk, 0, stream>>>(hB, ln1g + (long)l * D_,
                                                   ln1b + (long)l * D_, xB);
    gemm_bf16_kernel<EPI_QKV><<<gDD, blk, 0, stream>>>(
        xB, WqB + (long)l * D_ * D_, bq + (long)l * D_, nullptr, nullptr, qB, S_, D_, D_);
    gemm_bf16_kernel<EPI_QKV><<<gDD, blk, 0, stream>>>(
        xB, WkB + (long)l * D_ * D_, bk + (long)l * D_, nullptr, nullptr, kB, S_, D_, D_);
    gemm_bf16_kernel<EPI_QKV><<<gDD, blk, 0, stream>>>(
        xB, WvB + (long)l * D_ * D_, bv + (long)l * D_, nullptr, nullptr, vB, S_, D_, D_);
    attention_kernel<<<dim3(S_ / BQ, H_), blk, 0, stream>>>(qB, kB, vB, aoB);
    gemm_bf16_kernel<EPI_RESID><<<gDD, blk, 0, stream>>>(
        aoB, WoB + (long)l * D_ * D_, bo + (long)l * D_, hB, hB, nullptr, S_, D_, D_);
    layernorm_kernel<<<dim3(S_), blk, 0, stream>>>(hB, ln2g + (long)l * D_,
                                                   ln2b + (long)l * D_, xB);
    gemm_bf16_kernel<EPI_GELU><<<gDF, blk, 0, stream>>>(
        xB, W1B + (long)l * D_ * F_, b1 + (long)l * F_, nullptr, nullptr, ffB, S_, F_, D_);
    gemm_bf16_kernel<EPI_RESID><<<gDD, blk, 0, stream>>>(
        ffB, W2B + (long)l * F_ * D_, b2 + (long)l * D_, hB, hB, nullptr, S_, D_, F_);
  }
  layernorm_kernel<<<dim3(S_), blk, 0, stream>>>(hB, fg, fb, xB);
  // tied LM head: logits = x @ tok_emb^T  (tok_emb is V x D == N x K already)
  gemm_bf16_kernel<EPI_F32><<<dim3(V_ / BN, S_ / BM), blk, 0, stream>>>(
      xB, tokB, nullptr, nullptr, logits, nullptr, S_, V_, D_);
}
